// Net_49091476194058
// MI455X (gfx1250) — matrix operand order repaired and checked by reading, not compiled, not run on hardware
//
#include <hip/hip_runtime.h>
#include <hip/hip_bf16.h>

typedef __attribute__((ext_vector_type(16))) _Float16      v16h;
typedef __attribute__((ext_vector_type(8)))  float         v8f;
typedef __attribute__((ext_vector_type(8)))  unsigned int  v8u;

#define BATCH     262144
#define TILES32   (BATCH / 32)          // 8192 double-tiles of 32 rows
#define WAVES_PB  8                     // 256 threads = 8 wave32
#define GRID      512

// ---- LDS layout (bytes) ------------------------------------------------
// f16 weights, zero-padded: W[l] is [Nrows][stride] with stride = Kpad+2
//  L1: 96x194  @f16 off 0      (18624)
//  L2: 64x98   @f16 off 18624  (6272)
//  L3: 32x66   @f16 off 24896  (2112)
//  L4..L7: 32x34 @f16 off 27008/28096/29184/30272 (1088 each)
//  L8: 16x34  @f16 off 31360  (544)   -> 31904 f16 = 63808 B
// f32 biases  @ byte 63808, 336 floats (1344 B)
// act scratch @ byte 65152: per wave 2 tiles x (16x98 f16 = 1568 f16) x 8 waves
#define BIAS_BYTE  63808
#define ACT_BYTE   65152
#define SMEM_BYTES (65152 + 8 * 2 * 1568 * 2)   // 115328

template<int IN, int OUT, int NR, int STR>
__device__ __forceinline__ void stage(_Float16* wdst, float* bdst,
                                      const float* Wsrc, const float* Bsrc, int tid) {
  for (int idx = tid; idx < NR * STR; idx += 256) {
    const int r = idx / STR;
    const int c = idx - r * STR;
    float v = 0.0f;
    if (r < OUT && c < IN) v = Wsrc[r * IN + c];
    wdst[idx] = (_Float16)v;
  }
  for (int idx = tid; idx < NR; idx += 256)
    bdst[idx] = (idx < OUT) ? Bsrc[idx] : 0.0f;
}

// Load A fragments (16x32 f16 tiles) from activation scratch.
// ISA A layout: VGPR n, lane-half h holds K = 16*(n>>2) + 2*(n&3) + 8*h (+32*kt)
template<int KT>
__device__ __forceinline__ void load_a(v16h* a, const _Float16* act, int m, int h) {
#pragma unroll
  for (int kt = 0; kt < KT; ++kt) {
    v8u t;
#pragma unroll
    for (int n = 0; n < 8; ++n) {
      const int k = 16 * (n >> 2) + 2 * (n & 3) + 8 * h + 32 * kt;
      t[n] = *(const unsigned int*)(act + m * 98 + k);
    }
    a[kt] = __builtin_bit_cast(v16h, t);
  }
}

// One layer for TWO 16-row tiles sharing each B fragment:
// D = A * W^T (+bias, ReLU) via v_wmma_f32_16x16x32_f16.
// ISA B layout: VGPR n, lane-half h holds K = 16*(n>>2) + 2*(n&3) + 8*h (+32*kt), N = lane%16.
template<int KT, int NT, int SW, bool LAST>
__device__ __forceinline__ void compute2(const v16h* a0, const v16h* a1,
                                         const _Float16* wl, const float* bl,
                                         _Float16* act0, _Float16* act1,
                                         int m, int h, float* outp,
                                         long orow0, long orow1) {
#pragma unroll
  for (int nt = 0; nt < NT; ++nt) {
    v8f d0 = {0.f, 0.f, 0.f, 0.f, 0.f, 0.f, 0.f, 0.f};
    v8f d1 = {0.f, 0.f, 0.f, 0.f, 0.f, 0.f, 0.f, 0.f};
#pragma unroll
    for (int kt = 0; kt < KT; ++kt) {
      v8u t;
#pragma unroll
      for (int n = 0; n < 8; ++n) {
        const int k = 16 * (n >> 2) + 2 * (n & 3) + 8 * h + 32 * kt;
        t[n] = *(const unsigned int*)(wl + (nt * 16 + m) * SW + k);
      }
      const v16h bh = __builtin_bit_cast(v16h, t);
      d0 = __builtin_amdgcn_wmma_f32_16x16x32_f16(false, a0[kt], false, bh,
                                                  (short)0, d0, false, false);
      d1 = __builtin_amdgcn_wmma_f32_16x16x32_f16(false, a1[kt], false, bh,
                                                  (short)0, d1, false, false);
    }
    const float bias = bl[nt * 16 + m];
    if constexpr (LAST) {
      if (m == 0) {
#pragma unroll
        for (int v = 0; v < 8; ++v) {
          const float r0 = d0[v] + bias;
          const float r1 = d1[v] + bias;
          outp[orow0 + v + 8 * h] = 1.0f / (1.0f + __expf(-r0));
          outp[orow1 + v + 8 * h] = 1.0f / (1.0f + __expf(-r1));
        }
      }
    } else {
#pragma unroll
      for (int v = 0; v < 8; ++v) {
        float r0 = fmaxf(d0[v] + bias, 0.0f);
        float r1 = fmaxf(d1[v] + bias, 0.0f);
        act0[(v + 8 * h) * 98 + nt * 16 + m] = (_Float16)r0;
        act1[(v + 8 * h) * 98 + nt * 16 + m] = (_Float16)r1;
      }
    }
  }
}

__global__ __launch_bounds__(256, 1) void mlp8_wmma_kernel(
    const float* __restrict__ x,
    const float* __restrict__ w1, const float* __restrict__ b1,
    const float* __restrict__ w2, const float* __restrict__ b2,
    const float* __restrict__ w3, const float* __restrict__ b3,
    const float* __restrict__ w4, const float* __restrict__ b4,
    const float* __restrict__ w5, const float* __restrict__ b5,
    const float* __restrict__ w6, const float* __restrict__ b6,
    const float* __restrict__ w7, const float* __restrict__ b7,
    const float* __restrict__ w8, const float* __restrict__ b8,
    float* __restrict__ out) {
  extern __shared__ char smem[];
  _Float16* wlds   = (_Float16*)smem;
  float*    blds   = (float*)(smem + BIAS_BYTE);
  _Float16* actall = (_Float16*)(smem + ACT_BYTE);

  const int tid = threadIdx.x;
  stage<168, 84, 96, 194>(wlds,         blds,       w1, b1, tid);
  stage< 84, 42, 64,  98>(wlds + 18624, blds +  96, w2, b2, tid);
  stage< 42, 21, 32,  66>(wlds + 24896, blds + 160, w3, b3, tid);
  stage< 21, 12, 32,  34>(wlds + 27008, blds + 192, w4, b4, tid);
  stage< 12, 10, 32,  34>(wlds + 28096, blds + 224, w5, b5, tid);
  stage< 10,  5, 32,  34>(wlds + 29184, blds + 256, w6, b6, tid);
  stage<  5,  2, 32,  34>(wlds + 30272, blds + 288, w7, b7, tid);
  stage<  2,  1, 16,  34>(wlds + 31360, blds + 320, w8, b8, tid);
  __syncthreads();

  const int wave = tid >> 5;        // wave32
  const int lane = tid & 31;
  const int h    = lane >> 4;       // lane half
  const int m    = lane & 15;       // row-in-tile / N index
  _Float16* act0 = actall + wave * (2 * 1568);
  _Float16* act1 = act0 + 1568;

  const int nwaves = gridDim.x * WAVES_PB;
  for (int tl = blockIdx.x * WAVES_PB + wave; tl < TILES32; tl += nwaves) {
    const long row0 = (long)tl * 32 + m;
    const long row1 = row0 + 16;

    // Prefetch next double-tile's rows (global_prefetch_b8).
    if (tl + nwaves < TILES32) {
      const float* np = x + ((long)(tl + nwaves) * 32 + m) * 168;
      __builtin_prefetch(np, 0, 1);
      __builtin_prefetch(np + 96, 0, 1);
      __builtin_prefetch(np + 16 * 168, 0, 1);
      __builtin_prefetch(np + 16 * 168 + 96, 0, 1);
    }

    // ---- Layer 1: A fragments straight from global x (f32 -> f16) ----
    v16h a0[6], a1[6];
#pragma unroll
    for (int kt = 0; kt < 6; ++kt) {
#pragma unroll
      for (int n = 0; n < 8; ++n) {
        const int col = 16 * (n >> 2) + 2 * (n & 3) + 8 * h + 32 * kt;
        float2 v0 = {0.0f, 0.0f};
        float2 v1 = {0.0f, 0.0f};
        if (col < 168) {
          v0 = *(const float2*)(x + row0 * 168 + col);
          v1 = *(const float2*)(x + row1 * 168 + col);
        }
        a0[kt][2 * n]     = (_Float16)v0.x;
        a0[kt][2 * n + 1] = (_Float16)v0.y;
        a1[kt][2 * n]     = (_Float16)v1.x;
        a1[kt][2 * n + 1] = (_Float16)v1.y;
      }
    }
    compute2<6, 6, 194, false>(a0, a1, wlds,         blds,       act0, act1, m, h, out, 0, 0);

    load_a<3>(a0, act0, m, h); load_a<3>(a1, act1, m, h);
    compute2<3, 4,  98, false>(a0, a1, wlds + 18624, blds +  96, act0, act1, m, h, out, 0, 0);
    load_a<2>(a0, act0, m, h); load_a<2>(a1, act1, m, h);
    compute2<2, 2,  66, false>(a0, a1, wlds + 24896, blds + 160, act0, act1, m, h, out, 0, 0);
    load_a<1>(a0, act0, m, h); load_a<1>(a1, act1, m, h);
    compute2<1, 2,  34, false>(a0, a1, wlds + 27008, blds + 192, act0, act1, m, h, out, 0, 0);
    load_a<1>(a0, act0, m, h); load_a<1>(a1, act1, m, h);
    compute2<1, 2,  34, false>(a0, a1, wlds + 28096, blds + 224, act0, act1, m, h, out, 0, 0);
    load_a<1>(a0, act0, m, h); load_a<1>(a1, act1, m, h);
    compute2<1, 2,  34, false>(a0, a1, wlds + 29184, blds + 256, act0, act1, m, h, out, 0, 0);
    load_a<1>(a0, act0, m, h); load_a<1>(a1, act1, m, h);
    compute2<1, 2,  34, false>(a0, a1, wlds + 30272, blds + 288, act0, act1, m, h, out, 0, 0);
    load_a<1>(a0, act0, m, h); load_a<1>(a1, act1, m, h);
    compute2<1, 1,  34, true >(a0, a1, wlds + 31360, blds + 320, act0, act1, m, h, out,
                               (long)tl * 32, (long)tl * 32 + 16);
  }
}

extern "C" void kernel_launch(void* const* d_in, const int* in_sizes, int n_in,
                              void* d_out, int out_size, void* d_ws, size_t ws_size,
                              hipStream_t stream) {
  (void)in_sizes; (void)n_in; (void)out_size; (void)d_ws; (void)ws_size;
  const float* x = (const float*)d_in[0];
  const float* W[8];
  const float* B[8];
  for (int i = 0; i < 8; ++i) {
    W[i] = (const float*)d_in[1 + 2 * i];
    B[i] = (const float*)d_in[2 + 2 * i];
  }
  float* out = (float*)d_out;

  (void)hipFuncSetAttribute((const void*)mlp8_wmma_kernel,
                            hipFuncAttributeMaxDynamicSharedMemorySize,
                            SMEM_BYTES);

  hipLaunchKernelGGL(mlp8_wmma_kernel, dim3(GRID), dim3(256), SMEM_BYTES, stream,
                     x, W[0], B[0], W[1], B[1], W[2], B[2], W[3], B[3],
                     W[4], B[4], W[5], B[5], W[6], B[6], W[7], B[7], out);
}
